// ProbTraffic_gin_25134148616282
// MI455X (gfx1250) — matrix operand order repaired and checked by reading, not compiled, not run on hardware
//
#include <hip/hip_runtime.h>
#include <hip/hip_bf16.h>

// ---------------------------------------------------------------------------
// GIN inference for MI455X (gfx1250, wave32, WMMA).
//   T:(N,3) f32, src/dst:(E,) i32, params = pytree leaves (jax sorted-key order)
// Pipeline:
//   deg -> [per layer: scatter-mean agg -> x=h+agg -> WMMA GEMM w/ fused
//   BN+ReLU epilogues] -> column pooling -> 1-block head (LN + SELU MLP).
// GEMM: weights staged to LDS via global_load_async_to_lds_b128 (ASYNCcnt),
// converted/transposed to f16 so B fragments are contiguous ds_load_b128,
// A fragments loaded as float4; 8x v_wmma_f32_16x16x32_f16 per wave.
// Output: concat(mu, mu, logvar) = 384 f32.
// ---------------------------------------------------------------------------

typedef __attribute__((ext_vector_type(16))) _Float16 v16h;
typedef __attribute__((ext_vector_type(8)))  float    v8f;

#define BN_EPS 1e-5f
#define LN_EPS 1e-5f

// ---------------- utility kernels ----------------

__global__ void fill_zero(float* __restrict__ p, long n) {
    long i = (long)blockIdx.x * blockDim.x + threadIdx.x;
    if (i < n) p[i] = 0.0f;
}

__global__ void count_deg(const int* __restrict__ dst, float* __restrict__ deg, int E) {
    int e = blockIdx.x * blockDim.x + threadIdx.x;
    if (e < E) atomicAdd(&deg[dst[e]], 1.0f);
}

__global__ void inv_deg(float* __restrict__ deg, int n) {
    int i = blockIdx.x * blockDim.x + threadIdx.x;
    if (i < n) deg[i] = 1.0f / fmaxf(deg[i], 1.0f);
}

// scatter h[src] into agg[dst] (float atomics; working set lives in 192MB L2)
__global__ void scatter3(const float* __restrict__ H, const int* __restrict__ src,
                         const int* __restrict__ dst, float* __restrict__ agg, int E) {
    int e = blockIdx.x * blockDim.x + threadIdx.x;
    if (e >= E) return;
    int s = src[e], d = dst[e];
    atomicAdd(&agg[d * 3 + 0], H[s * 3 + 0]);
    atomicAdd(&agg[d * 3 + 1], H[s * 3 + 1]);
    atomicAdd(&agg[d * 3 + 2], H[s * 3 + 2]);
}

__global__ void scatter64(const float* __restrict__ H, const int* __restrict__ src,
                          const int* __restrict__ dst, float* __restrict__ agg, int E) {
    int t = blockIdx.x * blockDim.x + threadIdx.x;
    int e = t >> 4;                 // 16 threads per edge, 4 feats each
    if (e >= E) return;
    int f = (t & 15) * 4;
    int s = src[e], d = dst[e];
    float4 v = *(const float4*)(H + (size_t)s * 64 + f);
    atomicAdd(&agg[(size_t)d * 64 + f + 0], v.x);
    atomicAdd(&agg[(size_t)d * 64 + f + 1], v.y);
    atomicAdd(&agg[(size_t)d * 64 + f + 2], v.z);
    atomicAdd(&agg[(size_t)d * 64 + f + 3], v.w);
}

// in-place: agg = h + agg * deg_inv[node]   (GIN eps=0 + mean aggregation)
__global__ void combine(float* __restrict__ agg, const float* __restrict__ H,
                        const float* __restrict__ deginv, long total, int dim) {
    long i = (long)blockIdx.x * blockDim.x + threadIdx.x;
    if (i >= total) return;
    long node = i / dim;
    agg[i] = H[i] + agg[i] * deginv[node];
}

// fold linear bias + batchnorm(s) into per-channel scale/shift:
//   stage1: v = relu(acc*s1 + t1)    s1=g1/sqrt(v1+eps), t1=b1+(lb-m1)*s1
//   stage2: v = relu(v*s2 + t2)      (identity s2=1,t2=0 when only one BN)
__global__ void prep_ep(const float* __restrict__ lb,
                        const float* __restrict__ g1, const float* __restrict__ b1,
                        const float* __restrict__ m1, const float* __restrict__ v1,
                        const float* __restrict__ g2, const float* __restrict__ b2,
                        const float* __restrict__ m2, const float* __restrict__ v2,
                        float* __restrict__ ep, int dim, int has2) {
    int i = blockIdx.x * blockDim.x + threadIdx.x;
    if (i >= dim) return;
    float s1 = g1[i] * rsqrtf(v1[i] + BN_EPS);
    ep[i]       = s1;
    ep[dim + i] = b1[i] + (lb[i] - m1[i]) * s1;
    if (has2) {
        float s2 = g2[i] * rsqrtf(v2[i] + BN_EPS);
        ep[2 * dim + i] = s2;
        ep[3 * dim + i] = b2[i] - m2[i] * s2;
    } else {
        ep[2 * dim + i] = 1.0f;
        ep[3 * dim + i] = 0.0f;
    }
}

// layer-0 lin1: K=3 too small for WMMA -> scalar kernel, same fused epilogue
__global__ void lin_in3(const float* __restrict__ X, const float* __restrict__ W,
                        const float* __restrict__ ep, float* __restrict__ Y, long total) {
    long i = (long)blockIdx.x * blockDim.x + threadIdx.x;
    if (i >= total) return;
    long n = i >> 6;
    int  o = (int)(i & 63);
    float acc = X[n * 3 + 0] * W[0 * 64 + o]
              + X[n * 3 + 1] * W[1 * 64 + o]
              + X[n * 3 + 2] * W[2 * 64 + o];
    float v = fmaxf(acc * ep[o] + ep[64 + o], 0.0f);
    v = fmaxf(v * ep[128 + o] + ep[192 + o], 0.0f);
    Y[i] = v;
}

// ---------------- WMMA GEMM: Y = epilogue( X(Mx64) @ W(64x64) ) ----------------
// Block = 128 threads = 4 waves; wave w computes M-tile (blockIdx.x*4+w) x all
// 64 output columns. K=64 -> 2x v_wmma_f32_16x16x32_f16 per output tile,
// 8 WMMA per wave.
// Weight staging: f32 W async-copied to LDS (global_load_async_to_lds_b128,
// ASYNCcnt), then converted to f16 *transposed* (wT[col*64+k]) so each lane's
// 16 B-fragment halves are 32 contiguous bytes -> 2x ds_load_b128.
__global__ void gemm_wmma64(const float* __restrict__ X, const float* __restrict__ W,
                            const float* __restrict__ ep, float* __restrict__ Y, int M) {
    __shared__ __align__(16) float    wstage[64 * 64];   // 16KB raw f32 (async staged)
    __shared__ __align__(16) _Float16 wT[64 * 64];       // 8KB f16, transposed

    int tid = threadIdx.x;

    // async stage: 4096 floats = 1024 x 16B chunks; 128 threads x 8 chunks
    {
        unsigned ldsbase = (unsigned)(unsigned long long)(void*)wstage;
#pragma unroll
        for (int j = 0; j < 8; ++j) {
            int c = tid + 128 * j;                 // chunk index
            unsigned       loff = ldsbase + (unsigned)c * 16u;
            const float*   gp   = W + (size_t)c * 4;
            asm volatile("global_load_async_to_lds_b128 %0, %1, off"
                         :: "v"(loff), "v"(gp) : "memory");
        }
        asm volatile("s_wait_asynccnt 0" ::: "memory");
    }
    __syncthreads();

    // convert + transpose to f16: wT[col*64 + k] = (f16)W[k*64 + col]
    for (int i = tid; i < 64 * 64; i += 128) {
        int k = i >> 6, col = i & 63;
        wT[col * 64 + k] = (_Float16)wstage[i];
    }
    __syncthreads();

    int wave = tid >> 5;
    int lane = tid & 31;
    int mtile = blockIdx.x * 4 + wave;
    if (mtile * 16 >= M) return;          // wave-uniform: EXEC stays all-ones below

    int hi = lane >> 4;                   // half-wave select
    int ln = lane & 15;

    // B fragments (this code's contiguous K order, K=32): lane -> column n; lanes 0-15 hold
    // K=kb..kb+15, lanes 16-31 hold K=kb+16..kb+31, halves sequential in K.
    // Contiguous in wT -> vector LDS loads.
    v16h bfrag[4][2];
#pragma unroll
    for (int nt = 0; nt < 4; ++nt)
#pragma unroll
        for (int kh = 0; kh < 2; ++kh) {
            int kb  = kh * 32 + hi * 16;
            int col = nt * 16 + ln;
            bfrag[nt][kh] = *(const v16h*)(wT + col * 64 + kb);
        }

    // A fragments (16-bit A 16x32, this code's contiguous K order): lane -> row m; per 32-K chunk,
    // lanes 0-15 hold K 0-15, lanes 16-31 hold K 16-31 (matches B).
    // Each half = 16 contiguous f32 -> float4 loads.
    const float* xrow = X + ((size_t)mtile * 16 + ln) * 64;
    v16h afrag[2];
#pragma unroll
    for (int kh = 0; kh < 2; ++kh) {
        int kb = kh * 32 + hi * 16;
        float4 p0 = *(const float4*)(xrow + kb);
        float4 p1 = *(const float4*)(xrow + kb + 4);
        float4 p2 = *(const float4*)(xrow + kb + 8);
        float4 p3 = *(const float4*)(xrow + kb + 12);
        afrag[kh][0]  = (_Float16)p0.x;  afrag[kh][1]  = (_Float16)p0.y;
        afrag[kh][2]  = (_Float16)p0.z;  afrag[kh][3]  = (_Float16)p0.w;
        afrag[kh][4]  = (_Float16)p1.x;  afrag[kh][5]  = (_Float16)p1.y;
        afrag[kh][6]  = (_Float16)p1.z;  afrag[kh][7]  = (_Float16)p1.w;
        afrag[kh][8]  = (_Float16)p2.x;  afrag[kh][9]  = (_Float16)p2.y;
        afrag[kh][10] = (_Float16)p2.z;  afrag[kh][11] = (_Float16)p2.w;
        afrag[kh][12] = (_Float16)p3.x;  afrag[kh][13] = (_Float16)p3.y;
        afrag[kh][14] = (_Float16)p3.z;  afrag[kh][15] = (_Float16)p3.w;
    }

    v8f acc[4] = {};
#pragma unroll
    for (int nt = 0; nt < 4; ++nt) {
        acc[nt] = __builtin_amdgcn_wmma_f32_16x16x32_f16(
            false, afrag[0], false, bfrag[nt][0], (short)0, acc[nt], false, false);
        acc[nt] = __builtin_amdgcn_wmma_f32_16x16x32_f16(
            false, afrag[1], false, bfrag[nt][1], (short)0, acc[nt], false, false);
    }

    // C/D layout: VGPR r -> row (r + 8*hi), lane%16 -> column. Fused epilogue:
    // two scale/shift+ReLU stages (BN folds precomputed in ep[]).
#pragma unroll
    for (int nt = 0; nt < 4; ++nt) {
        int col = nt * 16 + ln;
        float s1 = ep[col], t1 = ep[64 + col];
        float s2 = ep[128 + col], t2 = ep[192 + col];
#pragma unroll
        for (int r = 0; r < 8; ++r) {
            int m = mtile * 16 + r + hi * 8;
            float v = acc[nt][r];
            v = fmaxf(v * s1 + t1, 0.0f);
            v = fmaxf(v * s2 + t2, 0.0f);
            Y[(size_t)m * 64 + col] = v;
        }
    }
}

// column-wise sum pooling: one block per column
__global__ void pool_sum(const float* __restrict__ H, float* __restrict__ out,
                         int n, int dim) {
    __shared__ float red[256];
    int col = blockIdx.x;
    int t = threadIdx.x;
    float acc = 0.0f;
    for (int r = t; r < n; r += 256) acc += H[(size_t)r * dim + col];
    red[t] = acc;
    __syncthreads();
    for (int s = 128; s > 0; s >>= 1) {
        if (t < s) red[t] += red[t + s];
        __syncthreads();
    }
    if (t == 0) out[col] = red[0];
}

// single-block head: prediction heads + LayerNorm(128) + SELU MLP -> (mu,mu,logvar)
__global__ void head_kernel(const float* __restrict__ pooled,   // [0..2]p0 [3..66]p1 [67..130]p2
                            const float* __restrict__ W0, const float* __restrict__ b0,
                            const float* __restrict__ W1, const float* __restrict__ b1,
                            const float* __restrict__ W2, const float* __restrict__ b2,
                            const float* __restrict__ lng, const float* __restrict__ lnb,
                            const float* __restrict__ Wf1, const float* __restrict__ bf1,
                            const float* __restrict__ W21, const float* __restrict__ b21,
                            const float* __restrict__ W22, const float* __restrict__ b22,
                            float* __restrict__ out) {
    __shared__ float sc[128], cn[128], h1v[256], red[256];
    int t = threadIdx.x;

    if (t < 128) {
        float a = b0[t] + b1[t] + b2[t];
        for (int k = 0; k < 3; ++k)  a += pooled[k]      * W0[k * 128 + t];
        for (int k = 0; k < 64; ++k) a += pooled[3 + k]  * W1[k * 128 + t];
        for (int k = 0; k < 64; ++k) a += pooled[67 + k] * W2[k * 128 + t];
        sc[t] = a;
    }
    __syncthreads();

    red[t] = (t < 128) ? sc[t] : 0.0f;
    __syncthreads();
    for (int s = 128; s > 0; s >>= 1) { if (t < s) red[t] += red[t + s]; __syncthreads(); }
    float mean = red[0] / 128.0f;
    __syncthreads();

    red[t] = (t < 128) ? (sc[t] - mean) * (sc[t] - mean) : 0.0f;
    __syncthreads();
    for (int s = 128; s > 0; s >>= 1) { if (t < s) red[t] += red[t + s]; __syncthreads(); }
    float rstd = rsqrtf(red[0] / 128.0f + LN_EPS);
    __syncthreads();

    if (t < 128) cn[t] = (sc[t] - mean) * rstd * lng[t] + lnb[t];
    __syncthreads();

    {   // fc1 + SELU (256 outputs, one per thread)
        float a = bf1[t];
        for (int o = 0; o < 128; ++o) a += cn[o] * Wf1[o * 256 + t];
        const float SC = 1.0507009873554805f, AL = 1.6732632423543772f;
        h1v[t] = a > 0.0f ? SC * a : SC * AL * (expf(a) - 1.0f);
    }
    __syncthreads();

    if (t < 128) {
        float mu = b21[t], lv = b22[t];
        for (int j = 0; j < 256; ++j) {
            mu += h1v[j] * W21[j * 128 + t];
            lv += h1v[j] * W22[j * 128 + t];
        }
        out[t]       = mu;
        out[128 + t] = mu;
        out[256 + t] = lv;
    }
}

// ---------------- launch ----------------

extern "C" void kernel_launch(void* const* d_in, const int* in_sizes, int n_in,
                              void* d_out, int out_size, void* d_ws, size_t ws_size,
                              hipStream_t stream) {
    (void)n_in; (void)out_size; (void)ws_size;
#define FIN(i) ((const float*)d_in[i])
    const float* T   = FIN(0);
    const int*   src = (const int*)d_in[1];
    const int*   dst = (const int*)d_in[2];
    const int N = in_sizes[0] / 3;
    const int E = in_sizes[1];

    // params flattened in jax pytree order (dict keys sorted at each level):
    // convs[0]: 3:bn1.b 4:bn1.g 5:bn1.m 6:bn1.v  7:bna.b 8:g 9:m 10:v
    //           11:bno.b 12:g 13:m 14:v  15:lin1.W 16:lin1.b 17:lin2.W 18:lin2.b
    // convs[1]: 19..22 bn1  23..26 bna  27..30 bno  31:lin1.W 32:b 33:lin2.W 34:b
    // 35:fc1.W 36:fc1.b  37:fc21.W 38:b  39:fc22.W 40:b  41:ln.b 42:ln.g
    // preds: 43:W0 44:b0  45:W1 46:b1  47:W2 48:b2

    float* ws = (float*)d_ws;
    size_t o = 0;
    float* deg    = ws + o; o += (size_t)((N + 63) / 64) * 64;
    float* agg    = ws + o; o += (size_t)N * 64;       // also x; reused as h2
    float* y1     = ws + o; o += (size_t)N * 64;
    float* h1     = ws + o; o += (size_t)N * 64;
    float* pooled = ws + o; o += 192;
    float* ep0 = ws + o; o += 256;   // layer0 lin1
    float* ep1 = ws + o; o += 256;   // layer0 lin2 (double BN)
    float* ep2 = ws + o; o += 256;   // layer1 lin1
    float* ep3 = ws + o; o += 256;   // layer1 lin2 (double BN)
    float* h2 = agg;

    const int TPB = 256;
    auto blocks = [](long n, int t) { return (int)((n + t - 1) / t); };

    // degrees -> deg_inv
    fill_zero<<<blocks(N, TPB), TPB, 0, stream>>>(deg, N);
    count_deg<<<blocks(E, TPB), TPB, 0, stream>>>(dst, deg, E);
    inv_deg<<<blocks(N, TPB), TPB, 0, stream>>>(deg, N);

    // fold bias+BN into epilogue scale/shift tables
    prep_ep<<<1, 64, 0, stream>>>(FIN(16), FIN(4), FIN(3), FIN(5), FIN(6),
                                  FIN(4), FIN(3), FIN(5), FIN(6), ep0, 64, 0);
    prep_ep<<<1, 64, 0, stream>>>(FIN(18), FIN(8), FIN(7), FIN(9), FIN(10),
                                  FIN(12), FIN(11), FIN(13), FIN(14), ep1, 64, 1);
    prep_ep<<<1, 64, 0, stream>>>(FIN(32), FIN(20), FIN(19), FIN(21), FIN(22),
                                  FIN(20), FIN(19), FIN(21), FIN(22), ep2, 64, 0);
    prep_ep<<<1, 64, 0, stream>>>(FIN(34), FIN(24), FIN(23), FIN(25), FIN(26),
                                  FIN(28), FIN(27), FIN(29), FIN(30), ep3, 64, 1);

    const int gemmBlocks = blocks((long)(N / 16), 4);   // 4 waves (M-tiles) per block

    // ---- layer 0 (in_dim=3) ----
    fill_zero<<<blocks((long)N * 3, TPB), TPB, 0, stream>>>(agg, (long)N * 3);
    scatter3<<<blocks(E, TPB), TPB, 0, stream>>>(T, src, dst, agg, E);
    combine<<<blocks((long)N * 3, TPB), TPB, 0, stream>>>(agg, T, deg, (long)N * 3, 3);
    lin_in3<<<blocks((long)N * 64, TPB), TPB, 0, stream>>>(agg, FIN(15), ep0, y1, (long)N * 64);
    gemm_wmma64<<<gemmBlocks, 128, 0, stream>>>(y1, FIN(17), ep1, h1, N);

    // ---- layer 1 (in_dim=64) ----
    fill_zero<<<blocks((long)N * 64, TPB), TPB, 0, stream>>>(agg, (long)N * 64);
    scatter64<<<blocks((long)E * 16, TPB), TPB, 0, stream>>>(h1, src, dst, agg, E);
    combine<<<blocks((long)N * 64, TPB), TPB, 0, stream>>>(agg, h1, deg, (long)N * 64, 64);
    gemm_wmma64<<<gemmBlocks, 128, 0, stream>>>(agg, FIN(31), ep2, y1, N);
    gemm_wmma64<<<gemmBlocks, 128, 0, stream>>>(y1, FIN(33), ep3, h2, N);

    // ---- pooling + head ----
    pool_sum<<<3,  TPB, 0, stream>>>(T,  pooled + 0,  N, 3);
    pool_sum<<<64, TPB, 0, stream>>>(h1, pooled + 3,  N, 64);
    pool_sum<<<64, TPB, 0, stream>>>(h2, pooled + 67, N, 64);

    head_kernel<<<1, 256, 0, stream>>>(pooled,
                                       FIN(43), FIN(44), FIN(45), FIN(46), FIN(47), FIN(48),
                                       FIN(42), FIN(41),
                                       FIN(35), FIN(36), FIN(37), FIN(38), FIN(39), FIN(40),
                                       (float*)d_out);
#undef FIN
}
